// Prunus_56873956933742
// MI455X (gfx1250) — matrix operand order repaired and checked by reading, not compiled, not run on hardware
//
#include <hip/hip_runtime.h>
#include <math.h>
#include <stdint.h>

// ---------------------------------------------------------------------------
// Types for gfx1250 WMMA / async staging
// ---------------------------------------------------------------------------
typedef __attribute__((ext_vector_type(16))) __bf16          v16bf;
typedef __attribute__((ext_vector_type(8)))  float           v8f;
typedef __attribute__((ext_vector_type(16))) unsigned short  v16us;
typedef __attribute__((ext_vector_type(8)))  unsigned short  v8us;
typedef __attribute__((ext_vector_type(4)))  unsigned short  v4us;
typedef __attribute__((ext_vector_type(4)))  int             v4i;

#if __has_builtin(__builtin_amdgcn_global_load_async_to_lds_b128)
#define HAVE_ASYNC_LDS 1
typedef v4i __attribute__((address_space(1)))* g_v4i_ptr;   // global int4*
typedef v4i __attribute__((address_space(3)))* l_v4i_ptr;   // LDS int4*
#else
#define HAVE_ASYNC_LDS 0
#endif

#if HAVE_ASYNC_LDS
static __device__ __forceinline__ void async_copy_b128(const void* g, void* l) {
    // builtin prototype (probe-confirmed arg0): (int4 AS1*, int4 AS3*, imm offset, imm cpol)
    __builtin_amdgcn_global_load_async_to_lds_b128(
        (g_v4i_ptr)(void*)g, (l_v4i_ptr)l, 0, 0);
}
static __device__ __forceinline__ void async_wait_6() {
#if __has_builtin(__builtin_amdgcn_s_wait_asynccnt)
    __builtin_amdgcn_s_wait_asynccnt(6);
#else
    asm volatile("s_wait_asynccnt 6" ::: "memory");
#endif
}
static __device__ __forceinline__ void async_wait_0() {
#if __has_builtin(__builtin_amdgcn_s_wait_asynccnt)
    __builtin_amdgcn_s_wait_asynccnt(0);
#else
    asm volatile("s_wait_asynccnt 0" ::: "memory");
#endif
}
#endif

static __device__ __forceinline__ unsigned short f32_to_bf16(float f) {
    unsigned int u   = __float_as_uint(f);
    unsigned int lsb = (u >> 16) & 1u;
    u += 0x7fffu + lsb;            // round-to-nearest-even
    return (unsigned short)(u >> 16);
}
static __device__ __forceinline__ float bf16_to_f32(unsigned short s) {
    return __uint_as_float(((unsigned int)s) << 16);
}

// ---------------------------------------------------------------------------
// One-time f32 -> bf16 conversion (x and all GEMM weights), vectorized x4.
// ---------------------------------------------------------------------------
__global__ __launch_bounds__(256)
void cvt_f32_to_bf16(const float4* __restrict__ in, v4us* __restrict__ out, int n4)
{
    int i = blockIdx.x * blockDim.x + threadIdx.x;
    int stride = gridDim.x * blockDim.x;
    for (; i < n4; i += stride) {
        float4 v = in[i];
        v4us o = { f32_to_bf16(v.x), f32_to_bf16(v.y),
                   f32_to_bf16(v.z), f32_to_bf16(v.w) };
        out[i] = o;
    }
}

// ---------------------------------------------------------------------------
// Tiled bf16 WMMA GEMM:  out = epilogue( A[M,K](bf16) @ W[N,K](bf16)^T + bias )
//   Block tile 128x256, K-step 32. 256 threads = 8 wave32 in 2(M)x4(N) grid;
//   each wave computes 64x64 = 4x4 v_wmma_f32_16x16x32_bf16 accumulators.
//   Double-buffered LDS; tiles staged with GLOBAL_LOAD_ASYNC_TO_LDS_B128
//   (ASYNCcnt) when available, overlapping tile k+1 DMA with tile k compute.
// ---------------------------------------------------------------------------
#define TBM 128
#define TBN 256
#define TBK 32

template <bool HAS_BN, bool DO_RELU, bool OUT_BF16>
__global__ __launch_bounds__(256)
void gemm_wmma(const unsigned short* __restrict__ A, const unsigned short* __restrict__ W,
               const float* __restrict__ bias, const float* __restrict__ gamma,
               const float* __restrict__ beta, float rs,
               void* __restrict__ Ov, int M, int N, int K, int ldo)
{
    __shared__ __align__(32) unsigned short As[2][TBM * TBK];
    __shared__ __align__(32) unsigned short Bs[2][TBN * TBK];

    const int tid    = threadIdx.x;
    const int lane   = tid & 31;
    const int wid    = tid >> 5;      // 0..7
    const int waveN  = wid & 3;       // 4 waves across N
    const int waveM  = wid >> 2;      // 2 waves across M
    const int blockM = blockIdx.y * TBM;
    const int blockN = blockIdx.x * TBN;
    const int khalf  = lane >> 4;     // 0/1
    const int lrow   = lane & 15;

    v8f acc[4][4] = {};

    const int nk = K / TBK;

    auto stage_tile = [&](int kt, int buf) {
        const long kb = (long)kt * TBK;
        // A tile: 128x32 bf16 = 8KB -> 2 x 16B chunks per thread
        #pragma unroll
        for (int c = 0; c < 2; ++c) {
            int id  = tid + 256 * c;          // 0..511
            int row = id >> 2;                // 0..127
            int ce  = (id & 3) * 8;           // element column
            const unsigned short* g = A + (long)(blockM + row) * K + kb + ce;
            unsigned short* l = &As[buf][row * TBK + ce];
#if HAVE_ASYNC_LDS
            async_copy_b128(g, l);
#else
            *(v8us*)l = *(const v8us*)g;
#endif
        }
        // W tile: 256x32 bf16 = 16KB -> 4 x 16B chunks per thread (N-guarded)
        #pragma unroll
        for (int c = 0; c < 4; ++c) {
            int id  = tid + 256 * c;          // 0..1023
            int row = id >> 2;                // 0..255
            int ce  = (id & 3) * 8;
            unsigned short* l = &Bs[buf][row * TBK + ce];
            int gn = blockN + row;
            if (gn < N) {
                const unsigned short* g = W + (long)gn * K + kb + ce;
#if HAVE_ASYNC_LDS
                async_copy_b128(g, l);
#else
                *(v8us*)l = *(const v8us*)g;
#endif
            } else {
                v8us z = {0, 0, 0, 0, 0, 0, 0, 0};
                *(v8us*)l = z;
            }
        }
    };

    stage_tile(0, 0);
    for (int kt = 0; kt < nk; ++kt) {
        const int buf = kt & 1;
        if (kt + 1 < nk) {
            stage_tile(kt + 1, buf ^ 1);      // DMA next tile while computing this one
#if HAVE_ASYNC_LDS
            async_wait_6();                    // in-order: drains tile kt's 6 copies
#endif
        } else {
#if HAVE_ASYNC_LDS
            async_wait_0();
#endif
        }
        __syncthreads();

        // B fragments: lane holds col n = lrow, 16 contiguous K at khalf*16
        union BF { v16us u; v16bf b; } bfr[4];
        #pragma unroll
        for (int nt = 0; nt < 4; ++nt) {
            int n = waveN * 64 + nt * 16 + lrow;
            bfr[nt].u = *(const v16us*)(&Bs[buf][n * TBK + khalf * 16]);
        }
        // A fragments + 16 WMMAs
        #pragma unroll
        for (int mt = 0; mt < 4; ++mt) {
            int m = waveM * 64 + mt * 16 + lrow;
            union AF { v16us u; struct { v8us lo, hi; } h; v16bf b; } af;
            af.h.lo = *(const v8us*)(&As[buf][m * TBK + khalf * 16]);      // K=khalf*16..+7
            af.h.hi = *(const v8us*)(&As[buf][m * TBK + khalf * 16 + 8]); // K=khalf*16+8..+15
            #pragma unroll
            for (int nt = 0; nt < 4; ++nt) {
                acc[mt][nt] = __builtin_amdgcn_wmma_f32_16x16x32_bf16(
                    false, af.b, false, bfr[nt].b, (short)0, acc[mt][nt], false, false);
            }
        }
        __syncthreads();
    }

    // ---- epilogue: bias (+BN) (+ReLU), store f32 or bf16 ----
    #pragma unroll
    for (int mt = 0; mt < 4; ++mt) {
        #pragma unroll
        for (int nt = 0; nt < 4; ++nt) {
            #pragma unroll
            for (int v = 0; v < 8; ++v) {
                int row = blockM + waveM * 64 + mt * 16 + khalf * 8 + v;
                int col = blockN + waveN * 64 + nt * 16 + lrow;
                if (col < N) {
                    float val = acc[mt][nt][v] + bias[col];
                    if (HAS_BN)  val = gamma[col] * (val * rs) + beta[col];
                    if (DO_RELU) val = fmaxf(val, 0.0f);
                    if (OUT_BF16)
                        ((unsigned short*)Ov)[(long)row * ldo + col] = f32_to_bf16(val);
                    else
                        ((float*)Ov)[(long)row * ldo + col] = val;
                }
            }
        }
    }
}

// ---------------------------------------------------------------------------
// Domain head (2 outputs) + routing head (8 outputs, relu -> argmax).
// One wave32 per row; lanes stride K, shuffle-reduce 10 partial sums.
// (Dirichlet sampling is PRNG noise; deterministic argmax of logits is used.)
// ---------------------------------------------------------------------------
__global__ __launch_bounds__(256)
void domain_route_kernel(const unsigned short* __restrict__ pen,  // [B,K] bf16
                         const float* __restrict__ W_dfc, const float* __restrict__ b_dfc,
                         const float* __restrict__ W_sw,  const float* __restrict__ b_sw,
                         float* __restrict__ dom_out,              // [B,2]
                         int* __restrict__ idx_out,                // [B] in d_out
                         int* __restrict__ idx_ws,                 // [B] scratch
                         int B, int K)
{
    const int lane = threadIdx.x & 31;
    const int wid  = threadIdx.x >> 5;
    const int r    = blockIdx.x * 8 + wid;
    if (r >= B) return;

    float s[10];
    #pragma unroll
    for (int i = 0; i < 10; ++i) s[i] = 0.0f;

    for (int k = lane; k < K; k += 32) {
        float x = bf16_to_f32(pen[(long)r * K + k]);
        s[0] += x * W_dfc[k];
        s[1] += x * W_dfc[K + k];
        #pragma unroll
        for (int e = 0; e < 8; ++e) s[2 + e] += x * W_sw[e * K + k];
    }
    #pragma unroll
    for (int i = 0; i < 10; ++i) {
        #pragma unroll
        for (int off = 16; off > 0; off >>= 1)
            s[i] += __shfl_xor(s[i], off);
    }
    if (lane == 0) {
        dom_out[r * 2 + 0] = s[0] + b_dfc[0];
        dom_out[r * 2 + 1] = s[1] + b_dfc[1];
        int best = 0; float bv = -1.0f;
        #pragma unroll
        for (int e = 0; e < 8; ++e) {
            float v = fmaxf(s[2 + e] + b_sw[e], 0.0f);
            if (v > bv) { bv = v; best = e; }
        }
        idx_out[r] = best;
        idx_ws[r]  = best;
    }
}

// ---------------------------------------------------------------------------
// Gathered expert second layer: out[r,c] = W2e[e,c,:] . h_all[r, e*PS:+PS] + b2e[e,c]
// One 128-thread block per row; hidden slice broadcast through LDS.
// ---------------------------------------------------------------------------
__global__ __launch_bounds__(128)
void expert_out_kernel(const unsigned short* __restrict__ h_all, // [B, E*PS] bf16
                       const float* __restrict__ W2e,            // [E,C,PS]
                       const float* __restrict__ b2e,            // [E,C]
                       const int* __restrict__ idx,               // [B]
                       float* __restrict__ out,                   // [B,C]
                       int C, int PS, int HL)
{
    __shared__ float h[128];
    const int r = blockIdx.x;
    const int t = threadIdx.x;
    const int e = idx[r];
    h[t] = bf16_to_f32(h_all[(long)r * HL + e * PS + t]);
    __syncthreads();
    if (t < C) {
        const float* w = W2e + ((long)(e * C + t)) * PS;
        float acc = b2e[e * C + t];
        #pragma unroll 4
        for (int p = 0; p < 128; ++p) acc += w[p] * h[p];
        out[(long)r * C + t] = acc;
    }
}

// ---------------------------------------------------------------------------
// Launcher
// ---------------------------------------------------------------------------
extern "C" void kernel_launch(void* const* d_in, const int* in_sizes, int n_in,
                              void* d_out, int out_size, void* d_ws, size_t ws_size,
                              hipStream_t stream)
{
    (void)in_sizes; (void)n_in; (void)out_size; (void)ws_size;
    const int B = 8192, F = 2048, D = 2048, PL = 1024, C = 100, PS = 128;
    const float RS = 1.0f / sqrtf(1.0f + 1e-5f);   // BN eval, mean=0 var=1

    const float* x      = (const float*)d_in[0];
    const float* W_pre  = (const float*)d_in[1];
    const float* b_pre  = (const float*)d_in[2];
    const float* g_pre  = (const float*)d_in[3];
    const float* be_pre = (const float*)d_in[4];
    const float* W_disc = (const float*)d_in[5];
    const float* b_disc = (const float*)d_in[6];
    const float* g_disc = (const float*)d_in[7];
    const float* be_disc= (const float*)d_in[8];
    const float* W_dfc  = (const float*)d_in[9];
    const float* b_dfc  = (const float*)d_in[10];
    const float* W_sw   = (const float*)d_in[11];
    const float* b_sw   = (const float*)d_in[12];
    const float* W1e    = (const float*)d_in[13];  // [E,PS,D] == [1024,2048] flat
    const float* b1e    = (const float*)d_in[14];  // [E,PS]   == [1024] flat
    const float* W2e    = (const float*)d_in[15];
    const float* b2e    = (const float*)d_in[16];
    const float* Wc1    = (const float*)d_in[17];
    const float* bc1    = (const float*)d_in[18];
    const float* g_c    = (const float*)d_in[19];
    const float* be_c   = (const float*)d_in[20];
    const float* Wc2    = (const float*)d_in[21];
    const float* bc2    = (const float*)d_in[22];

    // d_out: [class_part B*C][class B*C][domain B*2][idx B]
    float* out_part = (float*)d_out;
    float* out_cls  = out_part + (size_t)B * C;
    float* out_dom  = out_cls  + (size_t)B * C;
    int*   out_idx  = (int*)(out_dom + (size_t)B * 2);

    // workspace layout (all bf16 slabs 2-byte elems)
    uint8_t* ws = (uint8_t*)d_ws;
    size_t off = 0;
    unsigned short* feat   = (unsigned short*)(ws + off); off += (size_t)B * D * 2;
    unsigned short* pen    = (unsigned short*)(ws + off); off += (size_t)B * PL * 2;
    unsigned short* hall   = (unsigned short*)(ws + off); off += (size_t)B * PL * 2;
    int*            idxw   = (int*)(ws + off);            off += (size_t)B * sizeof(int);
    off = (off + 255) & ~(size_t)255;
    unsigned short* x_bf   = (unsigned short*)(ws + off); off += (size_t)B * F * 2;
    unsigned short* Wpre_b = (unsigned short*)(ws + off); off += (size_t)D * F * 2;
    unsigned short* Wdis_b = (unsigned short*)(ws + off); off += (size_t)PL * D * 2;
    unsigned short* W1e_b  = (unsigned short*)(ws + off); off += (size_t)PL * D * 2;
    unsigned short* Wc1_b  = (unsigned short*)(ws + off); off += (size_t)PL * D * 2;
    unsigned short* Wc2_b  = (unsigned short*)(ws + off); off += (size_t)C * PL * 2;

    // ---- one-time f32 -> bf16 conversions ----
    auto cvt = [&](const float* src, unsigned short* dst, long n) {
        int n4 = (int)(n / 4);
        int grid = (n4 + 255) / 256; if (grid > 8192) grid = 8192;
        cvt_f32_to_bf16<<<dim3(grid), dim3(256), 0, stream>>>(
            (const float4*)src, (v4us*)dst, n4);
    };
    cvt(x,      x_bf,   (long)B * F);
    cvt(W_pre,  Wpre_b, (long)D * F);
    cvt(W_disc, Wdis_b, (long)PL * D);
    cvt(W1e,    W1e_b,  (long)PL * D);
    cvt(Wc1,    Wc1_b,  (long)PL * D);
    cvt(Wc2,    Wc2_b,  (long)C * PL);

    dim3 blk(256);
    // 1) feature = relu(bn(x @ W_pre^T + b_pre))
    gemm_wmma<true, true, true><<<dim3(D / TBN, B / TBM), blk, 0, stream>>>(
        x_bf, Wpre_b, b_pre, g_pre, be_pre, RS, feat, B, D, F, D);
    // 2) domain_penul = relu(bn(feature @ W_disc^T + b_disc))
    gemm_wmma<true, true, true><<<dim3(PL / TBN, B / TBM), blk, 0, stream>>>(
        feat, Wdis_b, b_disc, g_disc, be_disc, RS, pen, B, PL, D, PL);
    // 3) domain head + routing argmax
    domain_route_kernel<<<dim3(B / 8), blk, 0, stream>>>(
        pen, W_dfc, b_dfc, W_sw, b_sw, out_dom, out_idx, idxw, B, PL);
    // 4) expert hidden for all experts: relu(feature @ W1e_flat^T + b1e_flat)
    gemm_wmma<false, true, true><<<dim3(PL / TBN, B / TBM), blk, 0, stream>>>(
        feat, W1e_b, b1e, nullptr, nullptr, RS, hall, B, PL, D, PL);
    // 5) gathered expert output -> class_output_partitioned
    expert_out_kernel<<<dim3(B), dim3(128), 0, stream>>>(
        hall, W2e, b2e, idxw, out_part, C, PS, PL);
    // 6) classifier hidden = relu(bn(feature @ Wc1^T + bc1))   (reuse pen slab)
    gemm_wmma<true, true, true><<<dim3(PL / TBN, B / TBM), blk, 0, stream>>>(
        feat, Wc1_b, bc1, g_c, be_c, RS, pen, B, PL, D, PL);
    // 7) class_output = hidden @ Wc2^T + bc2   (N=100, guarded; f32 out)
    gemm_wmma<false, false, false><<<dim3((C + TBN - 1) / TBN, B / TBM), blk, 0, stream>>>(
        pen, Wc2_b, bc2, nullptr, nullptr, RS, out_cls, B, C, PL, C);
}
